// Seq2SeqModel_6098853560588
// MI455X (gfx1250) — compile-verified, matrix operand order checked by reading, not run on hardware
//
#include <hip/hip_runtime.h>
#include <stdint.h>

typedef __bf16 bf16;
typedef __attribute__((ext_vector_type(16))) __bf16 v16bf;
typedef __attribute__((ext_vector_type(8)))  float  v8f;

#define DEVI __device__ __forceinline__

namespace {
constexpr int B = 32, S = 64, T = 64, E = 512, H = 512, V = 32000;
constexpr int H4  = 4 * H;    // 2048 encoder gate width
constexpr int DH  = 2 * H;    // 1024 decoder hidden
constexpr int DG  = 4 * DH;   // 4096 decoder gate width
constexpr int IND = DH + E;   // 1536 decoder LSTM input width
}

// ---------------------------------------------------------------------------
// WMMA helpers (CDNA5 16x16x32 bf16, wave32)
// A frag: lane<16 -> row=lane, K {0..7,16..23}; lane>=16 -> row=lane-16, K {8..15,24..31}
// B frag for C = A @ W^T (W is N x K row-major): identical pattern with lane = N.
// ---------------------------------------------------------------------------
DEVI v16bf load_frag(const bf16* p, int ld) {
  const int lane = threadIdx.x & 31;
  const bf16* q = p + (size_t)(lane & 15) * ld + ((lane & 16) ? 8 : 0);
  v16bf f;
#pragma unroll
  for (int i = 0; i < 8; ++i) f[i] = q[i];
#pragma unroll
  for (int i = 0; i < 8; ++i) f[8 + i] = q[16 + i];
  return f;
}

DEVI v8f wmma_bf16(v16bf a, v16bf b, v8f c) {
  return __builtin_amdgcn_wmma_f32_16x16x32_bf16(false, a, false, b, (short)0, c,
                                                 false, false);
}

DEVI float sigmoidf(float x) { return 1.0f / (1.0f + __expf(-x)); }

// ---------------------------------------------------------------------------
// Elementwise utility kernels
// ---------------------------------------------------------------------------
__global__ void fill_f32_kernel(float* p, float v, int n) {
  int i = blockIdx.x * 256 + threadIdx.x;
  if (i < n) p[i] = v;
}
__global__ void fill_bf16_kernel(bf16* p, float v, int n) {
  int i = blockIdx.x * 256 + threadIdx.x;
  if (i < n) p[i] = (bf16)v;
}
__global__ void cvt_bf16_kernel(const float* __restrict__ in, bf16* __restrict__ out, int n) {
  int i = blockIdx.x * 256 + threadIdx.x;
  if (i < n) out[i] = (bf16)in[i];
}
__global__ void addvec_kernel(const float* __restrict__ a, const float* __restrict__ b,
                              float* __restrict__ o, int n) {
  int i = blockIdx.x * 256 + threadIdx.x;
  if (i < n) o[i] = a[i] + b[i];
}

// Encoder embedding gather, time-major: X[(s*B + b)*E + e] = enc_emb[tok[b,s]][e]
__global__ void embed_enc_kernel(const int* __restrict__ tok, const float* __restrict__ emb,
                                 bf16* __restrict__ X) {
  int i = blockIdx.x * 256 + threadIdx.x;
  if (i >= S * B * E) return;
  int e = i % E;
  int sb = i / E;
  int b = sb % B;
  int s = sb / B;
  int tk = tok[b * S + s];
  X[i] = (bf16)emb[(size_t)tk * E + e];
}

// ---------------------------------------------------------------------------
// Generic WMMA GEMM: C(M x N, fp32, row stride ldc) = A(M x K, bf16) @ W^T + bias
// W is N x K row-major (weight layout). Block = 4 waves, each wave one 16x16 tile.
// Used for the large-M GEMMs (input projections, attention projection).
// ---------------------------------------------------------------------------
__global__ void gemm_bf16_kernel(const bf16* __restrict__ A, int lda,
                                 const bf16* __restrict__ W, int ldw,
                                 const float* __restrict__ bias,
                                 float* __restrict__ C, long long ldc,
                                 int M, int N, int K) {
  const int wave = threadIdx.x >> 5;
  const int lane = threadIdx.x & 31;
  const int n0 = (blockIdx.x * 4 + wave) * 16;
  const int m0 = blockIdx.y * 16;
  if (n0 >= N || m0 >= M) return;  // wave-uniform: EXEC stays all-ones for WMMA
  const bf16* Arow = A + (size_t)m0 * lda;
  const bf16* Wrow = W + (size_t)n0 * ldw;
  v8f acc = {};
  for (int k = 0; k < K; k += 32) {
    __builtin_prefetch(Wrow + k + 128, 0, 1);  // global_prefetch for next weight chunk
    v16bf af = load_frag(Arow + k, lda);
    v16bf bf = load_frag(Wrow + k, ldw);
    acc = wmma_bf16(af, bf, acc);
  }
  const int col = n0 + (lane & 15);
  const int rb = (lane & 16) ? 8 : 0;
  const float bv = bias ? bias[col] : 0.0f;
#pragma unroll
  for (int r = 0; r < 8; ++r)
    C[(long long)(m0 + rb + r) * ldc + col] = acc[r] + bv;
}

// ---------------------------------------------------------------------------
// Skinny WMMA GEMM for fixed M = 32 (batch): each wave computes BOTH M-tiles
// for one 16-wide N tile, so every weight fragment is loaded once and reused
// for two WMMAs. This halves L2 weight traffic on the dominant per-step
// logits GEMM (32 x 32000, K=1024, Wout re-streamed from L2 every step).
// grid.x = N/64, block = 128 (4 waves).
// ---------------------------------------------------------------------------
__global__ void gemm_skinny32_kernel(const bf16* __restrict__ A,   // (32, K)
                                     const bf16* __restrict__ W,   // (N, K)
                                     const float* __restrict__ bias,
                                     float* __restrict__ C, long long ldc,
                                     int N, int K) {
  const int wave = threadIdx.x >> 5;
  const int lane = threadIdx.x & 31;
  const int n0 = (blockIdx.x * 4 + wave) * 16;
  if (n0 >= N) return;  // wave-uniform
  const bf16* Wrow = W + (size_t)n0 * K;
  v8f acc0 = {}, acc1 = {};
  for (int k = 0; k < K; k += 32) {
    __builtin_prefetch(Wrow + k + 128, 0, 1);
    v16bf wf = load_frag(Wrow + k, K);
    v16bf a0 = load_frag(A + k, K);
    v16bf a1 = load_frag(A + (size_t)16 * K + k, K);
    acc0 = wmma_bf16(a0, wf, acc0);
    acc1 = wmma_bf16(a1, wf, acc1);
  }
  const int col = n0 + (lane & 15);
  const int rb = (lane & 16) ? 8 : 0;
  const float bv = bias ? bias[col] : 0.0f;
#pragma unroll
  for (int r = 0; r < 8; ++r) {
    C[(long long)(rb + r) * ldc + col] = acc0[r] + bv;
    C[(long long)(16 + rb + r) * ldc + col] = acc1[r] + bv;
  }
}

// ---------------------------------------------------------------------------
// Fused encoder LSTM cell step.
// grid.x = H/16; block = 128 (4 waves, one per gate i/f/g/o).
// pre: (B, 4H) fp32 = x@Wih^T + bih + bhh for this timestep (bias folded).
// ---------------------------------------------------------------------------
__global__ void lstm_enc_step_kernel(const float* __restrict__ pre,
                                     const bf16* __restrict__ hprev,  // (B, H)
                                     const bf16* __restrict__ Whh,    // (4H, H)
                                     float* __restrict__ cst,         // (B, H)
                                     bf16* __restrict__ hnext,        // (B, H)
                                     float* __restrict__ hout) {      // AH + s*DH + dir*H
  const int g = threadIdx.x >> 5;
  const int lane = threadIdx.x & 31;
  const int n0 = blockIdx.x * 16;
  const int wrow = g * H + n0;
  const int colL = lane & 15;
  const int rb = (lane & 16) ? 8 : 0;

  v8f acc0, acc1;
#pragma unroll
  for (int r = 0; r < 8; ++r) {
    acc0[r] = pre[(size_t)(rb + r) * H4 + wrow + colL];
    acc1[r] = pre[(size_t)(16 + rb + r) * H4 + wrow + colL];
  }
  const bf16* Wr = Whh + (size_t)wrow * H;
  for (int k = 0; k < H; k += 32) {
    v16bf bf = load_frag(Wr + k, H);
    v16bf a0 = load_frag(hprev + k, H);
    v16bf a1 = load_frag(hprev + (size_t)16 * H + k, H);
    acc0 = wmma_bf16(a0, bf, acc0);
    acc1 = wmma_bf16(a1, bf, acc1);
  }
  __shared__ float gbuf[4][32][16];
#pragma unroll
  for (int r = 0; r < 8; ++r) {
    gbuf[g][rb + r][colL] = acc0[r];
    gbuf[g][16 + rb + r][colL] = acc1[r];
  }
  __syncthreads();
  for (int idx = threadIdx.x; idx < B * 16; idx += blockDim.x) {
    const int b = idx >> 4, c = idx & 15;
    const float iv = sigmoidf(gbuf[0][b][c]);
    const float fv = sigmoidf(gbuf[1][b][c]);
    const float gv = tanhf(gbuf[2][b][c]);
    const float ov = sigmoidf(gbuf[3][b][c]);
    const int col = n0 + c;
    const float cp = cst[b * H + col];
    const float cn = fv * cp + iv * gv;
    const float hn = ov * tanhf(cn);
    cst[b * H + col] = cn;
    hnext[b * H + col] = (bf16)hn;
    hout[(size_t)b * (S * DH) + col] = hn;
  }
}

// ---------------------------------------------------------------------------
// Fused decoder LSTM cell step.
// grid.x = DH/16 = 64; block = 128 (4 waves, one per gate).
// g = inp(32x1536)@Wih_d^T + h(32x1024)@Whh_d^T + bias
// ---------------------------------------------------------------------------
__global__ void lstm_dec_step_kernel(const bf16* __restrict__ inp,   // (B, IND)
                                     const bf16* __restrict__ hprev, // (B, DH)
                                     const bf16* __restrict__ Wih,   // (DG, IND)
                                     const bf16* __restrict__ Whh,   // (DG, DH)
                                     const float* __restrict__ bsum, // (DG)
                                     float* __restrict__ cst,        // (B, DH)
                                     float* __restrict__ hf32,       // (B, DH)
                                     bf16* __restrict__ hnext) {     // (B, DH)
  const int g = threadIdx.x >> 5;
  const int lane = threadIdx.x & 31;
  const int n0 = blockIdx.x * 16;
  const int wrow = g * DH + n0;
  const int colL = lane & 15;
  const int rb = (lane & 16) ? 8 : 0;

  const float bv = bsum[wrow + colL];
  v8f acc0, acc1;
#pragma unroll
  for (int r = 0; r < 8; ++r) { acc0[r] = bv; acc1[r] = bv; }

  const bf16* Wr0 = Wih + (size_t)wrow * IND;
  for (int k = 0; k < IND; k += 32) {
    v16bf bf = load_frag(Wr0 + k, IND);
    v16bf a0 = load_frag(inp + k, IND);
    v16bf a1 = load_frag(inp + (size_t)16 * IND + k, IND);
    acc0 = wmma_bf16(a0, bf, acc0);
    acc1 = wmma_bf16(a1, bf, acc1);
  }
  const bf16* Wr1 = Whh + (size_t)wrow * DH;
  for (int k = 0; k < DH; k += 32) {
    v16bf bf = load_frag(Wr1 + k, DH);
    v16bf a0 = load_frag(hprev + k, DH);
    v16bf a1 = load_frag(hprev + (size_t)16 * DH + k, DH);
    acc0 = wmma_bf16(a0, bf, acc0);
    acc1 = wmma_bf16(a1, bf, acc1);
  }
  __shared__ float gbuf[4][32][16];
#pragma unroll
  for (int r = 0; r < 8; ++r) {
    gbuf[g][rb + r][colL] = acc0[r];
    gbuf[g][16 + rb + r][colL] = acc1[r];
  }
  __syncthreads();
  for (int idx = threadIdx.x; idx < B * 16; idx += blockDim.x) {
    const int b = idx >> 4, c = idx & 15;
    const float iv = sigmoidf(gbuf[0][b][c]);
    const float fv = sigmoidf(gbuf[1][b][c]);
    const float gv = tanhf(gbuf[2][b][c]);
    const float ov = sigmoidf(gbuf[3][b][c]);
    const int col = n0 + c;
    const float cp = cst[b * DH + col];
    const float cn = fv * cp + iv * gv;
    const float hn = ov * tanhf(cn);
    cst[b * DH + col] = cn;
    hf32[b * DH + col] = hn;
    hnext[b * DH + col] = (bf16)hn;
  }
}

// ---------------------------------------------------------------------------
// Attention: scores[b,s] = dot(A[b,s,:], h[b,:]), one wave per (b,s)
// ---------------------------------------------------------------------------
__global__ void attn_scores_kernel(const float* __restrict__ A, const float* __restrict__ h,
                                   float* __restrict__ scores) {
  const int gwave = (blockIdx.x * blockDim.x + threadIdx.x) >> 5;
  const int lane = threadIdx.x & 31;
  if (gwave >= B * S) return;
  const int b = gwave / S;
  const float* a = A + (size_t)gwave * DH;
  const float* hh = h + (size_t)b * DH;
  float sum = 0.0f;
  for (int k = lane; k < DH; k += 32) sum += a[k] * hh[k];
#pragma unroll
  for (int m = 16; m >= 1; m >>= 1) sum += __shfl_xor(sum, m, 32);
  if (lane == 0) scores[gwave] = sum;
}

// Softmax over S=64 per batch row. grid = B, block = 64.
__global__ void softmax_kernel(const float* __restrict__ scores, float* __restrict__ attn) {
  __shared__ float buf[64];
  const int b = blockIdx.x, s = threadIdx.x;
  const float v = scores[b * S + s];
  buf[s] = v;
  __syncthreads();
  for (int off = 32; off >= 1; off >>= 1) {
    if (s < off) buf[s] = fmaxf(buf[s], buf[s + off]);
    __syncthreads();
  }
  const float mx = buf[0];
  __syncthreads();
  const float e = __expf(v - mx);
  buf[s] = e;
  __syncthreads();
  for (int off = 32; off >= 1; off >>= 1) {
    if (s < off) buf[s] += buf[s + off];
    __syncthreads();
  }
  attn[b * S + s] = e / buf[0];
}

// Build decoder LSTM input: inp[b, 0:DH] = ctx, inp[b, DH:DH+E] = feed(t). bf16.
// grid = (B, IND/256), block = 256.
__global__ void build_inp_kernel(const float* __restrict__ attn, const float* __restrict__ AH,
                                 const int* __restrict__ target, const float* __restrict__ dec_emb,
                                 bf16* __restrict__ inp, int t) {
  const int b = blockIdx.x;
  const int col = blockIdx.y * 256 + threadIdx.x;
  float v;
  if (col < DH) {
    float s = 0.0f;
    const float* ah = AH + (size_t)b * S * DH + col;
    const float* at = attn + b * S;
#pragma unroll 4
    for (int i = 0; i < S; ++i) s += at[i] * ah[(size_t)i * DH];
    v = s;
  } else {
    v = (t == 0) ? 0.0f
                 : dec_emb[(size_t)target[b * T + (t - 1)] * E + (col - DH)];
  }
  inp[(size_t)b * IND + col] = (bf16)v;
}

// Decoder state init: dec_h = [hT_f | hT_b], dec_c = [cF | cB]
__global__ void dec_init_kernel(const float* __restrict__ AH, const float* __restrict__ cF,
                                const float* __restrict__ cB, float* __restrict__ decH,
                                bf16* __restrict__ decHbf, float* __restrict__ decC) {
  const int i = blockIdx.x * 256 + threadIdx.x;
  if (i >= B * DH) return;
  const int b = i >> 10, d = i & (DH - 1);
  float h;
  if (d < H) h = AH[(size_t)b * S * DH + (size_t)(S - 1) * DH + d];  // hT_f
  else       h = AH[(size_t)b * S * DH + d];                          // hT_b (stored at s=0, [H:2H])
  decH[i] = h;
  decHbf[i] = (bf16)h;
  decC[i] = (d < H) ? cF[b * H + d] : cB[b * H + (d - H)];
}

// Argmax over V per (b,t) row; first-max tie-break like jnp.argmax.
__global__ void argmax_kernel(const float* __restrict__ logits, float* __restrict__ out) {
  __shared__ float vbuf[256];
  __shared__ int ibuf[256];
  const int row = blockIdx.x;
  const float* p = logits + (size_t)row * V;
  float best = -__builtin_inff();
  int bi = 0;
  for (int v = threadIdx.x; v < V; v += 256)
    if (p[v] > best) { best = p[v]; bi = v; }
  vbuf[threadIdx.x] = best;
  ibuf[threadIdx.x] = bi;
  __syncthreads();
  for (int off = 128; off >= 1; off >>= 1) {
    if (threadIdx.x < off) {
      const float ov = vbuf[threadIdx.x + off];
      const int oi = ibuf[threadIdx.x + off];
      if (ov > vbuf[threadIdx.x] || (ov == vbuf[threadIdx.x] && oi < ibuf[threadIdx.x])) {
        vbuf[threadIdx.x] = ov;
        ibuf[threadIdx.x] = oi;
      }
    }
    __syncthreads();
  }
  if (threadIdx.x == 0) out[row] = (float)ibuf[0];
}

// ---------------------------------------------------------------------------
// Host orchestration
// ---------------------------------------------------------------------------
extern "C" void kernel_launch(void* const* d_in, const int* in_sizes, int n_in,
                              void* d_out, int out_size, void* d_ws, size_t ws_size,
                              hipStream_t stream) {
  (void)in_sizes; (void)n_in; (void)out_size; (void)ws_size;

  const int*   input_chunk  = (const int*)d_in[0];
  const int*   target_chunk = (const int*)d_in[1];
  const float* enc_emb = (const float*)d_in[2];
  const float* dec_emb = (const float*)d_in[3];
  const float* Wih_f = (const float*)d_in[4];
  const float* Whh_f = (const float*)d_in[5];
  const float* bih_f = (const float*)d_in[6];
  const float* bhh_f = (const float*)d_in[7];
  const float* Wih_b = (const float*)d_in[8];
  const float* Whh_b = (const float*)d_in[9];
  const float* bih_b = (const float*)d_in[10];
  const float* bhh_b = (const float*)d_in[11];
  const float* Wih_d = (const float*)d_in[12];
  const float* Whh_d = (const float*)d_in[13];
  const float* bih_d = (const float*)d_in[14];
  const float* bhh_d = (const float*)d_in[15];
  const float* Wa   = (const float*)d_in[16];
  const float* ba   = (const float*)d_in[17];
  const float* Wout = (const float*)d_in[18];
  const float* bout = (const float*)d_in[19];

  float* logits = (float*)d_out;                         // (B, T, V)
  float* outIdx = (float*)d_out + (size_t)B * T * V;     // (B, T) as float

  // Workspace bump allocator
  char* wp = (char*)d_ws;
  auto alloc = [&](size_t bytes) -> char* {
    char* r = (char*)(((uintptr_t)wp + 255) & ~(uintptr_t)255);
    wp = r + bytes;
    return r;
  };
  bf16* wihF = (bf16*)alloc(sizeof(bf16) * (size_t)H4 * E);
  bf16* whhF = (bf16*)alloc(sizeof(bf16) * (size_t)H4 * H);
  bf16* wihB = (bf16*)alloc(sizeof(bf16) * (size_t)H4 * E);
  bf16* whhB = (bf16*)alloc(sizeof(bf16) * (size_t)H4 * H);
  bf16* wihD = (bf16*)alloc(sizeof(bf16) * (size_t)DG * IND);
  bf16* whhD = (bf16*)alloc(sizeof(bf16) * (size_t)DG * DH);
  bf16* waBf = (bf16*)alloc(sizeof(bf16) * (size_t)DH * DH);
  bf16* woBf = (bf16*)alloc(sizeof(bf16) * (size_t)V * DH);
  bf16* Xbf  = (bf16*)alloc(sizeof(bf16) * (size_t)S * B * E);
  bf16* AHbf = (bf16*)alloc(sizeof(bf16) * (size_t)B * S * DH);
  bf16* hF[2] = {(bf16*)alloc(sizeof(bf16) * B * H), (bf16*)alloc(sizeof(bf16) * B * H)};
  bf16* hB[2] = {(bf16*)alloc(sizeof(bf16) * B * H), (bf16*)alloc(sizeof(bf16) * B * H)};
  bf16* decHbf[2] = {(bf16*)alloc(sizeof(bf16) * B * DH), (bf16*)alloc(sizeof(bf16) * B * DH)};
  bf16* inpBf = (bf16*)alloc(sizeof(bf16) * B * IND);

  float* preF = (float*)alloc(sizeof(float) * (size_t)S * B * H4);
  float* preB = (float*)alloc(sizeof(float) * (size_t)S * B * H4);
  float* AH   = (float*)alloc(sizeof(float) * (size_t)B * S * DH);
  float* Amat = (float*)alloc(sizeof(float) * (size_t)B * S * DH);
  float* cF   = (float*)alloc(sizeof(float) * B * H);
  float* cB   = (float*)alloc(sizeof(float) * B * H);
  float* decH = (float*)alloc(sizeof(float) * B * DH);
  float* decC = (float*)alloc(sizeof(float) * B * DH);
  float* scores = (float*)alloc(sizeof(float) * B * S);
  float* attn   = (float*)alloc(sizeof(float) * B * S);
  float* bsumF  = (float*)alloc(sizeof(float) * H4);
  float* bsumB  = (float*)alloc(sizeof(float) * H4);
  float* bsumD  = (float*)alloc(sizeof(float) * DG);

  auto cvt = [&](const float* src, bf16* dst, size_t n) {
    cvt_bf16_kernel<<<dim3((unsigned)((n + 255) / 256)), 256, 0, stream>>>(src, dst, (int)n);
  };

  // 1. Weight conversion + bias sums + zero state
  cvt(Wih_f, wihF, (size_t)H4 * E);
  cvt(Whh_f, whhF, (size_t)H4 * H);
  cvt(Wih_b, wihB, (size_t)H4 * E);
  cvt(Whh_b, whhB, (size_t)H4 * H);
  cvt(Wih_d, wihD, (size_t)DG * IND);
  cvt(Whh_d, whhD, (size_t)DG * DH);
  cvt(Wa, waBf, (size_t)DH * DH);
  cvt(Wout, woBf, (size_t)V * DH);
  addvec_kernel<<<dim3(H4 / 256), 256, 0, stream>>>(bih_f, bhh_f, bsumF, H4);
  addvec_kernel<<<dim3(H4 / 256), 256, 0, stream>>>(bih_b, bhh_b, bsumB, H4);
  addvec_kernel<<<dim3(DG / 256), 256, 0, stream>>>(bih_d, bhh_d, bsumD, DG);
  fill_f32_kernel<<<dim3(B * H / 256), 256, 0, stream>>>(cF, 0.0f, B * H);
  fill_f32_kernel<<<dim3(B * H / 256), 256, 0, stream>>>(cB, 0.0f, B * H);
  fill_bf16_kernel<<<dim3(B * H / 256), 256, 0, stream>>>(hF[0], 0.0f, B * H);
  fill_bf16_kernel<<<dim3(B * H / 256), 256, 0, stream>>>(hB[0], 0.0f, B * H);

  // 2. Embedding + input-projection GEMMs (bias folded)
  embed_enc_kernel<<<dim3((S * B * E) / 256), 256, 0, stream>>>(input_chunk, enc_emb, Xbf);
  gemm_bf16_kernel<<<dim3(H4 / 64, (S * B) / 16), 128, 0, stream>>>(
      Xbf, E, wihF, E, bsumF, preF, (long long)H4, S * B, H4, E);
  gemm_bf16_kernel<<<dim3(H4 / 64, (S * B) / 16), 128, 0, stream>>>(
      Xbf, E, wihB, E, bsumB, preB, (long long)H4, S * B, H4, E);

  // 3. Encoder scans (sequential, one fused WMMA cell kernel per step)
  for (int s = 0; s < S; ++s) {
    lstm_enc_step_kernel<<<dim3(H / 16), 128, 0, stream>>>(
        preF + (size_t)s * B * H4, hF[s & 1], whhF, cF, hF[(s + 1) & 1],
        AH + (size_t)s * DH);
  }
  for (int i = 0; i < S; ++i) {
    const int s = S - 1 - i;
    lstm_enc_step_kernel<<<dim3(H / 16), 128, 0, stream>>>(
        preB + (size_t)s * B * H4, hB[i & 1], whhB, cB, hB[(i + 1) & 1],
        AH + (size_t)s * DH + H);
  }

  // 4. Attention projection A = AH @ Wa^T + ba
  cvt(AH, AHbf, (size_t)B * S * DH);
  gemm_bf16_kernel<<<dim3(DH / 64, (B * S) / 16), 128, 0, stream>>>(
      AHbf, DH, waBf, DH, ba, Amat, (long long)DH, B * S, DH, DH);

  // 5. Decoder
  dec_init_kernel<<<dim3((B * DH) / 256), 256, 0, stream>>>(AH, cF, cB, decH, decHbf[0], decC);
  for (int t = 0; t < T; ++t) {
    attn_scores_kernel<<<dim3((B * S * 32) / 256), 256, 0, stream>>>(Amat, decH, scores);
    softmax_kernel<<<dim3(B), 64, 0, stream>>>(scores, attn);
    build_inp_kernel<<<dim3(B, IND / 256), 256, 0, stream>>>(attn, AH, target_chunk, dec_emb,
                                                             inpBf, t);
    lstm_dec_step_kernel<<<dim3(DH / 16), 128, 0, stream>>>(
        inpBf, decHbf[t & 1], wihD, whhD, bsumD, decC, decH, decHbf[(t + 1) & 1]);
    // logits[:, t, :] = h @ Wout^T + bout  (rows are batch, row stride T*V)
    // Skinny GEMM: both M-tiles per wave -> Wout streamed from L2 once per step.
    gemm_skinny32_kernel<<<dim3(V / 64), 128, 0, stream>>>(
        decHbf[(t + 1) & 1], woBf, bout, logits + (size_t)t * V,
        (long long)T * V, V, DH);
  }

  // 6. Argmax over vocab
  argmax_kernel<<<dim3(B * T), 256, 0, stream>>>(logits, outIdx);
}
